// WindowAttention_10136122818722
// MI455X (gfx1250) — matrix operand order repaired and checked by reading, not compiled, not run on hardware
//
#include <hip/hip_runtime.h>

// ---------- problem constants ----------
#define S49   49
#define HEADS 16
#define HD    32
#define DIM   512
#define NQKV  1536
#define BWIN  4096
#define NW_   64
#define MTOT  (BWIN * S49)          // 200704 = 128 * 1568 exactly

typedef __attribute__((ext_vector_type(16))) _Float16 v16h;
typedef __attribute__((ext_vector_type(8)))  _Float16 v8h;
typedef __attribute__((ext_vector_type(8)))  float    v8f;

__device__ __forceinline__ v8f wmma_f16(v16h a, v16h b, v8f c) {
    return __builtin_amdgcn_wmma_f32_16x16x32_f16(false, a, false, b, (short)0, c, false, false);
}

// CDNA5 async DMA: global memory -> LDS, 16B per active lane, tracked by ASYNCcnt.
// LDS destination VGPR holds the 32-bit LDS offset (generic addr truncated: ISA 10.2).
__device__ __forceinline__ void async_load_b128(void* lds, const void* gptr) {
    unsigned ldsoff = (unsigned)(unsigned long long)lds;
    asm volatile("global_load_async_to_lds_b128 %0, %1, off"
                 :: "v"(ldsoff), "v"(gptr) : "memory");
}
__device__ __forceinline__ void wait_async0() {
    asm volatile("s_wait_asynccnt 0" ::: "memory");
}

__device__ __forceinline__ v16h cat16(v8h lo, v8h hi) {
    v16h r;
#pragma unroll
    for (int i = 0; i < 8; ++i) { r[i] = lo[i]; r[i + 8] = hi[i]; }
    return r;
}

// A fragment, 16x32 f16 (M x K). base points at row 0 of the 16-row tile.
// lane 0-15: M=lane, K = kb..kb+15, kb = (lane>>4)*16
template <int STRIDE>
__device__ __forceinline__ v16h load_a_frag(const _Float16* base, int lane) {
    int m = lane & 15, kb = (lane >> 4) * 16;
    const _Float16* row = base + m * STRIDE;
    v8h lo = *(const v8h*)(row + kb);
    v8h hi = *(const v8h*)(row + kb + 8);
    return cat16(lo, hi);
}

// B fragment, 32x16 f16 (K x N), supplied as N rows of K (i.e. B^T row-major).
// lane 0-15: N=lane, K=0..15 ; lanes 16-31: same N, K=16..31
template <int STRIDE>
__device__ __forceinline__ v16h load_b_frag(const _Float16* base, int lane) {
    int n = lane & 15, kh = (lane >> 4) * 16;
    const _Float16* row = base + n * STRIDE;
    v8h lo = *(const v8h*)(row + kh);
    v8h hi = *(const v8h*)(row + kh + 8);
    return cat16(lo, hi);
}

// ---------------- kernel 0: weight transpose/convert + bias gather ----------------
#define PREP_W1 (NQKV * DIM)              // 786432
#define PREP_W2 (DIM * DIM)               // 262144
#define PREP_B  (HEADS * S49 * S49)       // 38416
#define PREP_TOTAL (PREP_W1 + PREP_W2 + PREP_B)

__global__ void prep_kernel(const float* __restrict__ Wqkv, const float* __restrict__ Wproj,
                            const float* __restrict__ bias_table, const int* __restrict__ rel_index,
                            _Float16* __restrict__ wqkvT, _Float16* __restrict__ wprojT,
                            float* __restrict__ bias_full) {
    int idx = blockIdx.x * 256 + threadIdx.x;
    if (idx < PREP_W1) {
        int n = idx >> 9, k = idx & 511;                       // [N=1536][K=512]
        wqkvT[idx] = (_Float16)Wqkv[k * NQKV + n];
    } else if (idx < PREP_W1 + PREP_W2) {
        int t = idx - PREP_W1;
        int n = t >> 9, k = t & 511;                           // [N=512][K=512]
        wprojT[t] = (_Float16)Wproj[k * DIM + n];
    } else if (idx < PREP_TOTAL) {
        int t = idx - (PREP_W1 + PREP_W2);
        int h = t / (S49 * S49), r = t % (S49 * S49);
        bias_full[t] = bias_table[rel_index[r] * HEADS + h];
    }
}

// ---------------- kernel 1: QKV GEMM (x fp32 -> q/k/v f16, head-major) ----------------
#define BM 128
#define BN 64
#define BK 32
#define LDA (BK + 8)   // 40 halves, 80B rows (16B aligned)

__global__ __launch_bounds__(256) void qkv_gemm(const float* __restrict__ x,
                                                const _Float16* __restrict__ wqkvT,
                                                const float* __restrict__ bqkv,
                                                _Float16* __restrict__ qg,
                                                _Float16* __restrict__ kg,
                                                _Float16* __restrict__ vg) {
    __shared__ __align__(16) _Float16 As[BM][LDA];
    __shared__ __align__(16) _Float16 Bs[BN][LDA];
    const int tid = threadIdx.x;
    const int lane = tid & 31;
    const int wid = tid >> 5;
    const int wm = wid & 3, wn = wid >> 2;      // 4 x 2 waves -> 128 x 64 tile
    const int m0 = blockIdx.y * BM;
    const int n0 = blockIdx.x * BN;

    v8f acc[2][2];
#pragma unroll
    for (int i = 0; i < 2; ++i)
#pragma unroll
        for (int j = 0; j < 2; ++j) acc[i][j] = (v8f)(0.0f);

    for (int k0 = 0; k0 < DIM; k0 += BK) {
        // B tile (64x32 f16): async DMA straight into LDS, 1 x b128 per thread
        {
            int row = tid >> 2, c8 = (tid & 3) * 8;
            async_load_b128(&Bs[row][c8], wqkvT + (size_t)(n0 + row) * DIM + k0 + c8);
        }
        // A tile (fp32 -> f16 conversion, must pass through VALU): 4 float4 per thread
#pragma unroll
        for (int j = 0; j < 4; ++j) {
            int idx = tid + j * 256;
            int row = idx >> 3, c4 = (idx & 7) * 4;
            float4 f = *(const float4*)(x + (size_t)(m0 + row) * DIM + k0 + c4);
            As[row][c4 + 0] = (_Float16)f.x;
            As[row][c4 + 1] = (_Float16)f.y;
            As[row][c4 + 2] = (_Float16)f.z;
            As[row][c4 + 3] = (_Float16)f.w;
        }
        wait_async0();
        __syncthreads();

        v16h a0 = load_a_frag<LDA>(&As[wm * 32][0], lane);
        v16h a1 = load_a_frag<LDA>(&As[wm * 32 + 16][0], lane);
        v16h b0 = load_b_frag<LDA>(&Bs[wn * 32][0], lane);
        v16h b1 = load_b_frag<LDA>(&Bs[wn * 32 + 16][0], lane);
        acc[0][0] = wmma_f16(a0, b0, acc[0][0]);
        acc[0][1] = wmma_f16(a0, b1, acc[0][1]);
        acc[1][0] = wmma_f16(a1, b0, acc[1][0]);
        acc[1][1] = wmma_f16(a1, b1, acc[1][1]);
        __syncthreads();
    }

    const int lh = lane >> 4, jc = lane & 15;
    const float qscale = 0.17677669529663689f;   // 32^-0.5
#pragma unroll
    for (int tm = 0; tm < 2; ++tm) {
#pragma unroll
        for (int tn = 0; tn < 2; ++tn) {
            int n = n0 + wn * 32 + tn * 16 + jc;
            float bias = bqkv[n];
            int part = n >> 9;                 // 0=q 1=k 2=v
            int rem = n & 511;
            int h = rem >> 5, d = rem & 31;
            _Float16* dst = (part == 0) ? qg : ((part == 1) ? kg : vg);
            float scl = (part == 0) ? qscale : 1.0f;
#pragma unroll
            for (int r = 0; r < 8; ++r) {
                int m = m0 + wm * 32 + tm * 16 + r + 8 * lh;
                int win = m / S49, s = m - win * S49;
                float v = (acc[tm][tn][r] + bias) * scl;
                dst[(((size_t)(win * HEADS + h) * S49 + s) << 5) + d] = (_Float16)v;
            }
        }
    }
}

// ---------------- kernel 2: fused attention, one block per (window, head) ----------------
#define LQK 40   // f16 row stride for Qs/Ks/Vs
#define LPS 72   // f16 row stride for probs (144B, 16B-aligned)

__global__ __launch_bounds__(128) void attn_kernel(const _Float16* __restrict__ qg,
                                                   const _Float16* __restrict__ kg,
                                                   const _Float16* __restrict__ vg,
                                                   const float* __restrict__ bias_full,
                                                   const float* __restrict__ mask,
                                                   _Float16* __restrict__ attnout) {
    __shared__ __align__(16) _Float16 Qs[64][LQK];
    __shared__ __align__(16) _Float16 Ks[64][LQK];
    __shared__ __align__(16) _Float16 Vs[64][LQK];
    __shared__ float BMs[S49 * S49];
    __shared__ float Sc[64][65];
    __shared__ __align__(16) _Float16 Ps[64][LPS];

    const int tid = threadIdx.x;
    const int lane = tid & 31;
    const int mw = tid >> 5;              // wave id = M tile (0..3)
    const int h = blockIdx.x;
    const int w = blockIdx.y;
    const size_t base = (size_t)(w * HEADS + h) * S49 * HD;

    // q/k/v tiles: 49*32 halves = 196 x 16B chunks each -> async DMA to LDS
    for (int c = tid; c < 196; c += 128) {
        int r = c >> 2, c8 = (c & 3) * 8;
        async_load_b128(&Qs[r][c8], qg + base + c * 8);
        async_load_b128(&Ks[r][c8], kg + base + c * 8);
        async_load_b128(&Vs[r][c8], vg + base + c * 8);
    }
    // zero-pad rows 49..63 (vector DS stores, disjoint from async region)
    for (int c = tid; c < (64 - S49) * 4; c += 128) {
        int r = S49 + (c >> 2), c8 = (c & 3) * 8;
        v8h z = (v8h)((_Float16)0.0f);
        *(v8h*)&Qs[r][c8] = z; *(v8h*)&Ks[r][c8] = z; *(v8h*)&Vs[r][c8] = z;
    }
    {
        const float* bf = bias_full + h * (S49 * S49);
        const float* mk = mask + (w & (NW_ - 1)) * (S49 * S49);
        for (int idx = tid; idx < S49 * S49; idx += 128) BMs[idx] = bf[idx] + mk[idx];
    }
    wait_async0();
    __syncthreads();

    // scores: q (A, K=32=hd) x k^T (B): 4 N-tiles per wave
    v16h aq = load_a_frag<LQK>(&Qs[mw * 16][0], lane);
    v8f sc[4];
#pragma unroll
    for (int nt = 0; nt < 4; ++nt) {
        sc[nt] = (v8f)(0.0f);
        v16h bk = load_b_frag<LQK>(&Ks[nt * 16][0], lane);
        sc[nt] = wmma_f16(aq, bk, sc[nt]);
    }
    const int lh = lane >> 4, jc = lane & 15;
#pragma unroll
    for (int nt = 0; nt < 4; ++nt) {
#pragma unroll
        for (int r = 0; r < 8; ++r) {
            int i = mw * 16 + r + 8 * lh;
            int j = nt * 16 + jc;
            float v = sc[nt][r];
            v = (i < S49 && j < S49) ? (v + BMs[i * S49 + j]) : -1e30f;
            Sc[i][j] = v;
        }
    }
    __syncthreads();

    // row softmax -> Ps (f16, zero padded to 64x64)
    if (tid < 64) {
        int i = tid;
        if (i < S49) {
            float mx = -1e30f;
            for (int j = 0; j < S49; ++j) mx = fmaxf(mx, Sc[i][j]);
            float sum = 0.0f;
            for (int j = 0; j < S49; ++j) { float e = __expf(Sc[i][j] - mx); Sc[i][j] = e; sum += e; }
            float inv = 1.0f / sum;
            for (int j = 0; j < 64; ++j)
                Ps[i][j] = (j < S49) ? (_Float16)(Sc[i][j] * inv) : (_Float16)0.0f;
        } else {
            for (int j = 0; j < 64; ++j) Ps[i][j] = (_Float16)0.0f;
        }
    }
    __syncthreads();

    // out = P (64x64) x V (64x32): K split into 2 WMMAs, N split into 2 tiles
    v8f o[2];
    o[0] = (v8f)(0.0f); o[1] = (v8f)(0.0f);
#pragma unroll
    for (int ks = 0; ks < 2; ++ks) {
        v16h ap = load_a_frag<LPS>(&Ps[mw * 16][ks * 32], lane);
#pragma unroll
        for (int nt = 0; nt < 2; ++nt) {
            int d = nt * 16 + jc;
            int kh = lh * 16;
            v16h bv;
#pragma unroll
            for (int e = 0; e < 16; ++e) bv[e] = Vs[ks * 32 + kh + e][d];
            o[nt] = wmma_f16(ap, bv, o[nt]);
        }
    }
#pragma unroll
    for (int nt = 0; nt < 2; ++nt) {
#pragma unroll
        for (int r = 0; r < 8; ++r) {
            int i = mw * 16 + r + 8 * lh;
            if (i < S49)
                attnout[((size_t)w * S49 + i) * DIM + h * HD + nt * 16 + jc] = (_Float16)o[nt][r];
        }
    }
}

// ---------------- kernel 3: projection GEMM (f16 x f16 -> f32 out + bproj) ----------------
__global__ __launch_bounds__(256) void proj_gemm(const _Float16* __restrict__ Ain,
                                                 const _Float16* __restrict__ wprojT,
                                                 const float* __restrict__ bproj,
                                                 float* __restrict__ out) {
    __shared__ __align__(16) _Float16 As[BM][LDA];
    __shared__ __align__(16) _Float16 Bs[BN][LDA];
    const int tid = threadIdx.x;
    const int lane = tid & 31;
    const int wid = tid >> 5;
    const int wm = wid & 3, wn = wid >> 2;
    const int m0 = blockIdx.y * BM;
    const int n0 = blockIdx.x * BN;

    v8f acc[2][2];
#pragma unroll
    for (int i = 0; i < 2; ++i)
#pragma unroll
        for (int j = 0; j < 2; ++j) acc[i][j] = (v8f)(0.0f);

    for (int k0 = 0; k0 < DIM; k0 += BK) {
        // A tile (128x32 f16 = 512 x b128 chunks) + B tile (256 chunks): all async DMA
#pragma unroll
        for (int j = 0; j < 2; ++j) {
            int idx = tid + j * 256;
            int row = idx >> 2, c8 = (idx & 3) * 8;
            async_load_b128(&As[row][c8], Ain + (size_t)(m0 + row) * DIM + k0 + c8);
        }
        {
            int row = tid >> 2, c8 = (tid & 3) * 8;
            async_load_b128(&Bs[row][c8], wprojT + (size_t)(n0 + row) * DIM + k0 + c8);
        }
        wait_async0();
        __syncthreads();

        v16h a0 = load_a_frag<LDA>(&As[wm * 32][0], lane);
        v16h a1 = load_a_frag<LDA>(&As[wm * 32 + 16][0], lane);
        v16h b0 = load_b_frag<LDA>(&Bs[wn * 32][0], lane);
        v16h b1 = load_b_frag<LDA>(&Bs[wn * 32 + 16][0], lane);
        acc[0][0] = wmma_f16(a0, b0, acc[0][0]);
        acc[0][1] = wmma_f16(a0, b1, acc[0][1]);
        acc[1][0] = wmma_f16(a1, b0, acc[1][0]);
        acc[1][1] = wmma_f16(a1, b1, acc[1][1]);
        __syncthreads();
    }

    const int lh = lane >> 4, jc = lane & 15;
#pragma unroll
    for (int tm = 0; tm < 2; ++tm) {
#pragma unroll
        for (int tn = 0; tn < 2; ++tn) {
            int n = n0 + wn * 32 + tn * 16 + jc;
            float bias = bproj[n];
#pragma unroll
            for (int r = 0; r < 8; ++r) {
                int m = m0 + wm * 32 + tm * 16 + r + 8 * lh;
                out[(size_t)m * DIM + n] = acc[tm][tn][r] + bias;
            }
        }
    }
}

// ---------------- host launch ----------------
extern "C" void kernel_launch(void* const* d_in, const int* in_sizes, int n_in,
                              void* d_out, int out_size, void* d_ws, size_t ws_size,
                              hipStream_t stream) {
    const float* x          = (const float*)d_in[0];
    const float* mask       = (const float*)d_in[1];
    const float* Wqkv       = (const float*)d_in[2];
    const float* bqkv       = (const float*)d_in[3];
    const float* Wproj      = (const float*)d_in[4];
    const float* bproj      = (const float*)d_in[5];
    const float* bias_table = (const float*)d_in[6];
    const int*   rel_index  = (const int*)d_in[7];
    float* out = (float*)d_out;

    char* ws = (char*)d_ws;
    size_t off = 0;
    auto carve = [&](size_t bytes) -> void* {
        void* p = ws + off;
        off += (bytes + 255) & ~(size_t)255;
        return p;
    };
    _Float16* wqkvT     = (_Float16*)carve((size_t)NQKV * DIM * 2);
    _Float16* wprojT    = (_Float16*)carve((size_t)DIM * DIM * 2);
    float*    bias_full = (float*)carve((size_t)HEADS * S49 * S49 * 4);
    size_t headsz = (size_t)BWIN * HEADS * S49 * HD * 2;
    _Float16* qg = (_Float16*)carve(headsz);
    _Float16* kg = (_Float16*)carve(headsz);
    _Float16* vg = (_Float16*)carve(headsz);
    _Float16* attn_o = (_Float16*)carve((size_t)MTOT * DIM * 2);
    (void)in_sizes; (void)n_in; (void)out_size; (void)ws_size;

    prep_kernel<<<dim3((PREP_TOTAL + 255) / 256), dim3(256), 0, stream>>>(
        Wqkv, Wproj, bias_table, rel_index, wqkvT, wprojT, bias_full);

    qkv_gemm<<<dim3(NQKV / BN, MTOT / BM), dim3(256), 0, stream>>>(
        x, wqkvT, bqkv, qg, kg, vg);

    attn_kernel<<<dim3(HEADS, BWIN), dim3(128), 0, stream>>>(
        qg, kg, vg, bias_full, mask, attn_o);

    proj_gemm<<<dim3(DIM / BN, MTOT / BM), dim3(256), 0, stream>>>(
        attn_o, wprojT, bproj, out);
}
